// DCNBlock_18588618457396
// MI455X (gfx1250) — compile-verified, matrix operand order checked by reading, not run on hardware
//
#include <hip/hip_runtime.h>
#include <hip/hip_bf16.h>

// ---------------- problem constants ----------------
constexpr int Bc  = 4;
constexpr int Cc  = 384;
constexpr int Hc  = 56;
constexpr int Wc  = 56;
constexpr int HWc = Hc * Wc;          // 3136
constexpr int NTOK = Bc * HWc;        // 12544
constexpr int Gc  = 12;
constexpr int GCc = 32;               // == wave32 lanes
constexpr int KTAP = 9;               // 3x3
constexpr int OMRAW = Gc * KTAP * 3;  // 324
constexpr int OMPAD = 384;            // padded to multiple of 64
constexpr int HIDc = 4 * Cc;          // 1536
constexpr float EPSc = 1e-5f;

typedef __attribute__((ext_vector_type(8)))  _Float16 v8h_t;
typedef __attribute__((ext_vector_type(16))) _Float16 v16h_t;
typedef __attribute__((ext_vector_type(8)))  float    v8f_t;

// ---------------- fragment load (16-bit A/B 16x32 ISA layout) ----------------
// lane 0-15: row = lane, K-half 0 ; lane 16-31: row = lane-16, K-half 1
// v16h elements e=0..7  -> K = half*8 + e
//               e=8..15 -> K = 16 + half*8 + (e-8)
__device__ __forceinline__ v16h_t load_frag(const _Float16* __restrict__ base,
                                            int ldk, int row0, int kk, int lane) {
    int r  = row0 + (lane & 15);
    int hl = lane >> 4;
    const _Float16* p = base + (size_t)r * ldk + kk + hl * 8;
    v8h_t lo = *(const v8h_t*)(p);
    v8h_t hi = *(const v8h_t*)(p + 16);
    return __builtin_shufflevector(lo, hi, 0,1,2,3,4,5,6,7,8,9,10,11,12,13,14,15);
}

// ---------------- generic WMMA GEMM: out(M,N) = A(M,K) * Wt(N,K)^T ----------------
// Ping-pong pipelined over K (two disjoint fragment sets, K unrolled by 64;
// all K used here are multiples of 64). No register rotation -> no movs.
// MODE 0: store f16            MODE 1: store f32
// MODE 2: fp32 out[m,n] += v   (residual in place)
// MODE 3: exact GELU, store f16
// MODE 4: y = v + resid[m,n]; store fp32 NCHW (token m -> (b,hw), channel n)
template <int MODE>
__global__ __launch_bounds__(256) void gemm_kernel(
    const _Float16* __restrict__ A, const _Float16* __restrict__ Wt,
    const float* __restrict__ bias, int nbias,
    void* __restrict__ outp, const float* __restrict__ resid,
    int M, int N, int Kd)
{
    const int lane = threadIdx.x & 31;
    const int wv   = threadIdx.x >> 5;
    const int m0   = (blockIdx.y * 8 + wv) * 16;
    const int n0   = blockIdx.x * 64;

    v8f_t acc0 = {}, acc1 = {}, acc2 = {}, acc3 = {};

    // set 0 preload (kk = 0)
    v16h_t a0  = load_frag(A,  Kd, m0,      0, lane);
    v16h_t b00 = load_frag(Wt, Kd, n0,      0, lane);
    v16h_t b01 = load_frag(Wt, Kd, n0 + 16, 0, lane);
    v16h_t b02 = load_frag(Wt, Kd, n0 + 32, 0, lane);
    v16h_t b03 = load_frag(Wt, Kd, n0 + 48, 0, lane);

    for (int kk = 0; kk < Kd; kk += 64) {
        // set 1 loads for kk+32 (always in range: Kd % 64 == 0)
        v16h_t a1  = load_frag(A,  Kd, m0,      kk + 32, lane);
        v16h_t b10 = load_frag(Wt, Kd, n0,      kk + 32, lane);
        v16h_t b11 = load_frag(Wt, Kd, n0 + 16, kk + 32, lane);
        v16h_t b12 = load_frag(Wt, Kd, n0 + 32, kk + 32, lane);
        v16h_t b13 = load_frag(Wt, Kd, n0 + 48, kk + 32, lane);

        // compute on set 0 (loads for set 1 in flight)
        acc0 = __builtin_amdgcn_wmma_f32_16x16x32_f16(false, a0, false, b00, (short)0, acc0, false, false);
        acc1 = __builtin_amdgcn_wmma_f32_16x16x32_f16(false, a0, false, b01, (short)0, acc1, false, false);
        acc2 = __builtin_amdgcn_wmma_f32_16x16x32_f16(false, a0, false, b02, (short)0, acc2, false, false);
        acc3 = __builtin_amdgcn_wmma_f32_16x16x32_f16(false, a0, false, b03, (short)0, acc3, false, false);

        // set 0 reload for kk+64 (uniform branch; skipped on last iteration)
        if (kk + 64 < Kd) {
            a0  = load_frag(A,  Kd, m0,      kk + 64, lane);
            b00 = load_frag(Wt, Kd, n0,      kk + 64, lane);
            b01 = load_frag(Wt, Kd, n0 + 16, kk + 64, lane);
            b02 = load_frag(Wt, Kd, n0 + 32, kk + 64, lane);
            b03 = load_frag(Wt, Kd, n0 + 48, kk + 64, lane);
            if (kk + 128 < Kd)  // speculative prefetch of A two chunks ahead
                __builtin_prefetch((const void*)(A + (size_t)(m0 + (lane & 15)) * Kd + kk + 128), 0, 1);
        }

        // compute on set 1 (loads for set 0 in flight)
        acc0 = __builtin_amdgcn_wmma_f32_16x16x32_f16(false, a1, false, b10, (short)0, acc0, false, false);
        acc1 = __builtin_amdgcn_wmma_f32_16x16x32_f16(false, a1, false, b11, (short)0, acc1, false, false);
        acc2 = __builtin_amdgcn_wmma_f32_16x16x32_f16(false, a1, false, b12, (short)0, acc2, false, false);
        acc3 = __builtin_amdgcn_wmma_f32_16x16x32_f16(false, a1, false, b13, (short)0, acc3, false, false);
    }

    // D layout: lane L -> col = L&15; VGPR r -> row = r + 8*(L>>4)
    const int hl = lane >> 4;
    const int nl = lane & 15;
    v8f_t accs[4] = {acc0, acc1, acc2, acc3};
#pragma unroll
    for (int j = 0; j < 4; ++j) {
        const int n = n0 + j * 16 + nl;
        const float bv = (n < nbias) ? bias[n] : 0.f;
#pragma unroll
        for (int r = 0; r < 8; ++r) {
            const int m = m0 + r + 8 * hl;
            float v = accs[j][r] + bv;
            if constexpr (MODE == 0) {
                ((_Float16*)outp)[(size_t)m * N + n] = (_Float16)v;
            } else if constexpr (MODE == 1) {
                ((float*)outp)[(size_t)m * N + n] = v;
            } else if constexpr (MODE == 2) {
                float* o = (float*)outp;
                o[(size_t)m * N + n] += v;
            } else if constexpr (MODE == 3) {
                float gv = 0.5f * v * (1.0f + erff(v * 0.70710678118654752f));
                ((_Float16*)outp)[(size_t)m * N + n] = (_Float16)gv;
            } else {
                float y  = v + resid[(size_t)m * N + n];
                int bb = m / HWc;
                int hw = m - bb * HWc;
                ((float*)outp)[((size_t)bb * Cc + n) * HWc + hw] = y;
            }
        }
    }
}

// ---------------- LayerNorm (one wave per token, lane covers 12 channels) ----------------
template <bool NCHW, bool WRITE_XP>
__global__ __launch_bounds__(256) void ln_kernel(
    const float* __restrict__ src, const float* __restrict__ gamma,
    const float* __restrict__ beta, _Float16* __restrict__ out16,
    float* __restrict__ xp)
{
    const int lane = threadIdx.x & 31;
    const int wv   = threadIdx.x >> 5;
    const int t    = blockIdx.x * 8 + wv;
    const int bb   = t / HWc;
    const int hw   = t - bb * HWc;

    float vreg[12];
    float s = 0.f, s2 = 0.f;
#pragma unroll
    for (int i = 0; i < 12; ++i) {
        const int c = lane + 32 * i;
        float v = NCHW ? src[((size_t)bb * Cc + c) * HWc + hw]
                       : src[(size_t)t * Cc + c];
        vreg[i] = v; s += v; s2 += v * v;
    }
#pragma unroll
    for (int off = 16; off >= 1; off >>= 1) {
        s  += __shfl_xor(s,  off, 32);
        s2 += __shfl_xor(s2, off, 32);
    }
    const float mu  = s * (1.f / Cc);
    const float var = s2 * (1.f / Cc) - mu * mu;
    const float rs  = rsqrtf(var + EPSc);
#pragma unroll
    for (int i = 0; i < 12; ++i) {
        const int c = lane + 32 * i;
        float vn = (vreg[i] - mu) * rs * gamma[c] + beta[c];
        out16[(size_t)t * Cc + c] = (_Float16)vn;
        if (WRITE_XP) xp[(size_t)t * Cc + c] = vreg[i];
    }
}

// ---------------- DCNv4 bilinear sampling: one wave per (token, group), lane = channel ----------------
__device__ __forceinline__ float dsample(const _Float16* __restrict__ val16,
                                         int bb, int y, int x, int cch) {
    if ((unsigned)y >= (unsigned)Hc || (unsigned)x >= (unsigned)Wc) return 0.f;
    return (float)val16[((size_t)(bb * HWc + y * Wc + x)) * Cc + cch];
}

__global__ __launch_bounds__(256) void dcn_kernel(
    const _Float16* __restrict__ val16, const float* __restrict__ om,
    _Float16* __restrict__ dcn16)
{
    const int lane = threadIdx.x & 31;
    const int wave = (blockIdx.x * blockDim.x + threadIdx.x) >> 5;
    const int g = wave % Gc;
    const int t = wave / Gc;
    const int bb = t / HWc;
    const int hw = t - bb * HWc;
    const int iy = hw / Wc;
    const int ix = hw - iy * Wc;
    const int cch = g * GCc + lane;

    const float* omp = om + (size_t)t * OMPAD + g * 27;  // 18 offsets + 9 masks
    float acc = 0.f;
#pragma unroll
    for (int k = 0; k < KTAP; ++k) {
        const float ox = omp[2 * k];
        const float oy = omp[2 * k + 1];
        const float mk = omp[18 + k];
        const float px = (float)ix + (float)(k % 3 - 1) + ox;
        const float py = (float)iy + (float)(k / 3 - 1) + oy;
        const float x0f = floorf(px), y0f = floorf(py);
        const float tx = px - x0f, ty = py - y0f;
        const int x0 = (int)x0f, y0 = (int)y0f;
        const float v00 = dsample(val16, bb, y0,     x0,     cch);
        const float v01 = dsample(val16, bb, y0,     x0 + 1, cch);
        const float v10 = dsample(val16, bb, y0 + 1, x0,     cch);
        const float v11 = dsample(val16, bb, y0 + 1, x0 + 1, cch);
        const float s = (v00 * (1.f - tx) + v01 * tx) * (1.f - ty)
                      + (v10 * (1.f - tx) + v11 * tx) * ty;
        acc += s * mk;
    }
    dcn16[(size_t)t * Cc + cch] = (_Float16)acc;
}

// ---------------- weight packing: (K,Nsrc) fp32 -> (Ndst,K) f16, zero pad ----------------
__global__ void pack_w(const float* __restrict__ src, _Float16* __restrict__ dst,
                       int Kd, int Nsrc, int Ndst) {
    const int id = blockIdx.x * blockDim.x + threadIdx.x;
    if (id >= Ndst * Kd) return;
    const int n = id / Kd;
    const int k = id - n * Kd;
    dst[id] = (n < Nsrc) ? (_Float16)src[(size_t)k * Nsrc + n] : (_Float16)0.f;
}

__global__ void pad_bias(const float* __restrict__ src, float* __restrict__ dst,
                         int Nsrc, int Ndst) {
    const int n = blockIdx.x * blockDim.x + threadIdx.x;
    if (n < Ndst) dst[n] = (n < Nsrc) ? src[n] : 0.f;
}

// ---------------- host-side orchestration ----------------
extern "C" void kernel_launch(void* const* d_in, const int* in_sizes, int n_in,
                              void* d_out, int out_size, void* d_ws, size_t ws_size,
                              hipStream_t stream) {
    const float* x       = (const float*)d_in[0];
    const float* ln1_g   = (const float*)d_in[1];
    const float* ln1_b   = (const float*)d_in[2];
    const float* vproj_w = (const float*)d_in[3];
    const float* vproj_b = (const float*)d_in[4];
    const float* om_w    = (const float*)d_in[5];
    const float* om_b    = (const float*)d_in[6];
    const float* oproj_w = (const float*)d_in[7];
    const float* oproj_b = (const float*)d_in[8];
    const float* ln2_g   = (const float*)d_in[9];
    const float* ln2_b   = (const float*)d_in[10];
    const float* fc1_w   = (const float*)d_in[11];
    const float* fc1_b   = (const float*)d_in[12];
    const float* fc2_w   = (const float*)d_in[13];
    const float* fc2_b   = (const float*)d_in[14];
    float* out = (float*)d_out;

    // bump allocator over d_ws
    char* wsp = (char*)d_ws;
    auto alloc = [&](size_t bytes) {
        char* p = wsp;
        wsp += (bytes + 255) & ~(size_t)255;
        return (void*)p;
    };
    _Float16* tn16   = (_Float16*)alloc((size_t)NTOK * Cc * 2);
    _Float16* val16  = (_Float16*)alloc((size_t)NTOK * Cc * 2);
    _Float16* dcn16  = (_Float16*)alloc((size_t)NTOK * Cc * 2);
    _Float16* h16    = (_Float16*)alloc((size_t)NTOK * Cc * 2);
    _Float16* ff16   = (_Float16*)alloc((size_t)NTOK * HIDc * 2);
    float*    xp     = (float*)   alloc((size_t)NTOK * Cc * 4);
    float*    omb    = (float*)   alloc((size_t)NTOK * OMPAD * 4);
    _Float16* WtV    = (_Float16*)alloc((size_t)Cc * Cc * 2);
    _Float16* WtO    = (_Float16*)alloc((size_t)OMPAD * Cc * 2);
    _Float16* WtP    = (_Float16*)alloc((size_t)Cc * Cc * 2);
    _Float16* Wt1    = (_Float16*)alloc((size_t)HIDc * Cc * 2);
    _Float16* Wt2    = (_Float16*)alloc((size_t)Cc * HIDc * 2);
    float*    biasO  = (float*)   alloc((size_t)OMPAD * 4);

    // 1) pack weights (one-time per launch, tiny)
    pack_w<<<(Cc * Cc) / 256, 256, 0, stream>>>(vproj_w, WtV, Cc, Cc, Cc);
    pack_w<<<(OMPAD * Cc) / 256, 256, 0, stream>>>(om_w, WtO, Cc, OMRAW, OMPAD);
    pack_w<<<(Cc * Cc) / 256, 256, 0, stream>>>(oproj_w, WtP, Cc, Cc, Cc);
    pack_w<<<(HIDc * Cc) / 256, 256, 0, stream>>>(fc1_w, Wt1, Cc, HIDc, HIDc);
    pack_w<<<(Cc * HIDc) / 256, 256, 0, stream>>>(fc2_w, Wt2, HIDc, Cc, Cc);
    pad_bias<<<(OMPAD + 127) / 128, 128, 0, stream>>>(om_b, biasO, OMRAW, OMPAD);

    // 2) LN1 (NCHW gather) -> tn16 (f16) + xp residual (fp32, token-major)
    ln_kernel<true, true><<<NTOK / 8, 256, 0, stream>>>(x, ln1_g, ln1_b, tn16, xp);

    // 3) val = tn @ vproj + b  -> f16
    dim3 gv(Cc / 64, NTOK / 128);
    gemm_kernel<0><<<gv, 256, 0, stream>>>(tn16, WtV, vproj_b, Cc, val16, nullptr,
                                           NTOK, Cc, Cc);
    // 4) om = tn @ om_w + b (padded N=384) -> fp32
    dim3 go(OMPAD / 64, NTOK / 128);
    gemm_kernel<1><<<go, 256, 0, stream>>>(tn16, WtO, biasO, OMPAD, omb, nullptr,
                                           NTOK, OMPAD, Cc);
    // 5) DCNv4 sampling -> dcn16
    dcn_kernel<<<(NTOK * Gc * 32) / 256, 256, 0, stream>>>(val16, omb, dcn16);

    // 6) xp += dcn @ oproj + b  (residual in place)
    gemm_kernel<2><<<gv, 256, 0, stream>>>(dcn16, WtP, oproj_b, Cc, xp, nullptr,
                                           NTOK, Cc, Cc);
    // 7) LN2 (token-major) -> h16
    ln_kernel<false, false><<<NTOK / 8, 256, 0, stream>>>(xp, ln2_g, ln2_b, h16, nullptr);

    // 8) ff = gelu(h @ fc1 + b) -> f16
    dim3 g1(HIDc / 64, NTOK / 128);
    gemm_kernel<3><<<g1, 256, 0, stream>>>(h16, Wt1, fc1_b, HIDc, ff16, nullptr,
                                           NTOK, HIDc, Cc);
    // 9) out(NCHW) = xp + ff @ fc2 + b
    gemm_kernel<4><<<gv, 256, 0, stream>>>(ff16, Wt2, fc2_b, Cc, out, xp,
                                           NTOK, Cc, HIDc);
    (void)in_sizes; (void)n_in; (void)out_size; (void)ws_size;
}
